// MultiplexInductiveSmoother_78580721648086
// MI455X (gfx1250) — matrix operand order repaired and checked by reading, not compiled, not run on hardware
//
#include <hip/hip_runtime.h>
#include <hip/hip_bf16.h>
#include <math.h>

typedef __bf16 bf16_t;
typedef __attribute__((ext_vector_type(16))) __bf16 v16bf;
typedef __attribute__((ext_vector_type(8)))  float  v8f;

// ============================================================================
// WMMA fragment helpers (gfx1250, wave32, v_wmma_f32_16x16x32_bf16 layouts)
//   A (16xK=32), lane l: row m = m0+(l&15); halves 0..7 -> K = k0+8*(l>>4)+e,
//                halves 8..15 -> K = k0+16+8*(l>>4)+(e-8)
//   B (K=32x16), lane l: col n = n0+(l&15); halves e -> K same as for A above
//   C/D f32, elem i, lane l: m = m0+8*(l>>4)+i, n = n0+(l&15)
// All operands are pre-converted bf16 -> each fragment is two 16B loads.
// ============================================================================
union frag_u { v16bf v; uint4 q[2]; };

__device__ __forceinline__ v16bf load_frag(const __bf16* __restrict__ base,
                                           int ld, int row, int e1, int e2) {
  frag_u f;
  const __bf16* p = base + (size_t)row * ld;
  f.q[0] = *reinterpret_cast<const uint4*>(p + e1);   // 8 bf16
  f.q[1] = *reinterpret_cast<const uint4*>(p + e2);   // 8 bf16
  return f.v;
}

// Y[M,N] = act( X[M,K] @ W[N,K]^T + bias[n] + rowv[m]*bias2[n] )
// ACT: 0 = none, 1 = PReLU(alphap[0]), 2 = LeakyReLU(0.2)
// OBF: store bf16 (feeds next GEMM) instead of f32.
// PRECONDITION: M % 16 == 0 (true for all call sites: M = 20000) -> no
// divergent guard in the k-loop, EXEC stays all-1s for WMMA.
// One wave computes a 16x64 strip; K templated so the k-loop fully unrolls;
// fragments are double-buffered so next-step loads overlap current WMMAs.
template <int K, int ACT, bool OBF>
__global__ __launch_bounds__(256)
void gemm_k(const __bf16* __restrict__ X, int lda,
            const __bf16* __restrict__ W, int ldw,
            const float* __restrict__ bias,
            const float* __restrict__ rowv,
            const float* __restrict__ bias2,
            float* __restrict__ Yf, __bf16* __restrict__ Yb, int ldy,
            int M, int N, const float* __restrict__ alphap)
{
  const int lane    = threadIdx.x & 31;
  const int wid     = threadIdx.x >> 5;
  const int ngroups = N >> 6;                       // N is a multiple of 64
  const int gw      = blockIdx.x * (blockDim.x >> 5) + wid;
  const int mtiles  = (M + 15) >> 4;
  const int mtile   = gw / ngroups;
  const int ng      = gw - mtile * ngroups;
  if (mtile >= mtiles) return;                      // wave-uniform: EXEC all-1s
  const int m0   = mtile * 16;
  const int half = lane >> 4;
  const int r    = lane & 15;
  const int am   = m0 + r;                          // always < M (M % 16 == 0)
  const int bo   = 16 * half;                       // unused: B fragments use ao1/ao2 like A
  const int ao1  = 8 * half;                        // A-fragment k sub-offsets
  const int ao2  = 16 + 8 * half;
  const int nb0  = ng * 64 + r;

  v8f acc[4];
#pragma unroll
  for (int t = 0; t < 4; ++t)
#pragma unroll
    for (int i = 0; i < 8; ++i) acc[t][i] = 0.0f;

  // prologue: fragments for k0 = 0
  v16bf a  = load_frag(X, lda, am,       ao1, ao2);
  v16bf b0 = load_frag(W, ldw, nb0 + 0,  ao1, ao2);
  v16bf b1 = load_frag(W, ldw, nb0 + 16, ao1, ao2);
  v16bf b2 = load_frag(W, ldw, nb0 + 32, ao1, ao2);
  v16bf b3 = load_frag(W, ldw, nb0 + 48, ao1, ao2);

#pragma unroll
  for (int k0 = 0; k0 < K; k0 += 32) {
    // issue next k-step's loads before consuming current fragments
    v16bf an = a, bn0 = b0, bn1 = b1, bn2 = b2, bn3 = b3;
    if (k0 + 32 < K) {                              // folds at compile time
      const int k1 = k0 + 32;
      an  = load_frag(X, lda, am,       k1 + ao1, k1 + ao2);
      bn0 = load_frag(W, ldw, nb0 + 0,  k1 + ao1, k1 + ao2);
      bn1 = load_frag(W, ldw, nb0 + 16, k1 + ao1, k1 + ao2);
      bn2 = load_frag(W, ldw, nb0 + 32, k1 + ao1, k1 + ao2);
      bn3 = load_frag(W, ldw, nb0 + 48, k1 + ao1, k1 + ao2);
    }
    acc[0] = __builtin_amdgcn_wmma_f32_16x16x32_bf16(false, a, false, b0, (short)0, acc[0], false, false);
    acc[1] = __builtin_amdgcn_wmma_f32_16x16x32_bf16(false, a, false, b1, (short)0, acc[1], false, false);
    acc[2] = __builtin_amdgcn_wmma_f32_16x16x32_bf16(false, a, false, b2, (short)0, acc[2], false, false);
    acc[3] = __builtin_amdgcn_wmma_f32_16x16x32_bf16(false, a, false, b3, (short)0, acc[3], false, false);
    a = an; b0 = bn0; b1 = bn1; b2 = bn2; b3 = bn3;
  }

  const float alpha = (ACT == 1) ? alphap[0] : 0.2f;
#pragma unroll
  for (int t = 0; t < 4; ++t) {
    const int n = ng * 64 + t * 16 + r;
    const float bn = bias  ? bias[n]  : 0.0f;
    const float b2e = bias2 ? bias2[n] : 0.0f;
#pragma unroll
    for (int i = 0; i < 8; ++i) {
      const int m = m0 + 8 * half + i;
      if (m < M) {
        float v = acc[t][i] + bn;
        if (rowv) v += rowv[m] * b2e;
        if (ACT != 0) v = (v >= 0.0f) ? v : alpha * v;
        if (OBF) Yb[(size_t)m * ldy + n] = (bf16_t)v;
        else     Yf[(size_t)m * ldy + n] = v;
      }
    }
  }
}

// ============================================================================
// f32 -> bf16 conversion kernels (weights / features, once per launch)
// ============================================================================
__global__ void cvt_k(const float* __restrict__ in, __bf16* __restrict__ out, int n) {
  int i = blockIdx.x * blockDim.x + threadIdx.x;
  if (i < n) out[i] = (bf16_t)in[i];
}

// pack a column slice: out[r, 0:outc] = (bf16) in[r, offs : offs+outc], stride inc
__global__ void cvt_stride_k(const float* __restrict__ in, __bf16* __restrict__ out,
                             int rows, int inc, int offs, int outc) {
  int i = blockIdx.x * blockDim.x + threadIdx.x;
  if (i < rows * outc) {
    int r = i / outc, c = i - r * outc;
    out[i] = (bf16_t)in[(size_t)r * inc + offs + c];
  }
}

// ============================================================================
// Small / setup kernels
// ============================================================================
__global__ void setup_k(const float* __restrict__ target,
                        const float* fW1, const float* fb1, const float* fa1,
                        const float* fW2, const float* fb2,
                        const float* rW1, const float* rb1, const float* ra1,
                        const float* rW2, const float* rb2,
                        const float* qW, const float* qb,
                        const float* kW, const float* kb,
                        const float* lemb, const float* aW1, const float* ab1,
                        float* zt_g, float* qk_g, float* ck_g,
                        float* c0f_g, float* c0r_g)
{
  __shared__ float tf[256], h[256], z[256], zt[256], Qv[128];
  const int t = threadIdx.x;
  tf[t] = target[t];
  __syncthreads();
  float s = 0.0f;
  for (int i = 0; i < 256; ++i) s += fW1[t * 256 + i] * tf[i];
  s += fb1[t];
  float a = fa1[0];
  h[t] = (s >= 0.0f) ? s : a * s;
  __syncthreads();
  s = 0.0f;
  for (int i = 0; i < 256; ++i) s += fW2[t * 256 + i] * h[i];
  z[t] = s + fb2[t];
  __syncthreads();
  s = 0.0f;
  for (int i = 0; i < 256; ++i) s += rW1[t * 256 + i] * tf[i];
  s += rb1[t];
  a = ra1[0];
  h[t] = (s >= 0.0f) ? s : a * s;
  __syncthreads();
  s = 0.0f;
  for (int i = 0; i < 256; ++i) s += rW2[t * 256 + i] * h[i];
  zt[t] = 0.5f * (z[t] + (s + rb2[t]));
  zt_g[t] = zt[t];
  __syncthreads();
  if (t < 128) {
    s = qb[t];
    for (int i = 0; i < 256; ++i) s += qW[t * 256 + i] * zt[i];
    Qv[t] = s;
  }
  __syncthreads();
  if (t < 128) {
    s = 0.0f;
    for (int i = 0; i < 128; ++i) s += kW[i * 128 + t] * Qv[i];
    qk_g[t] = s;
  }
  if (t == 128) {
    s = 0.0f;
    for (int i = 0; i < 128; ++i) s += kb[i] * Qv[i];
    ck_g[0] = s;
  }
  if (t < 128) {
    float base = ab1[t];
    for (int i = 0; i < 256; ++i) base += aW1[t * 528 + i] * zt[i];
    float c0 = base, c1 = base;
    for (int j = 0; j < 16; ++j) {
      c0 += aW1[t * 528 + 512 + j] * lemb[j];
      c1 += aW1[t * 528 + 512 + j] * lemb[16 + j];
    }
    c0f_g[t] = c0;
    c0r_g[t] = c1;
  }
}

// dscore[j] = (drug_features[j] . qk + ck) / sqrt(128)   -- one wave per row
__global__ void dscore_k(const float* __restrict__ drug,
                         const float* __restrict__ qk,
                         const float* __restrict__ ckp,
                         float* __restrict__ dscore, int ND)
{
  const int lane = threadIdx.x & 31;
  const int w = blockIdx.x * (blockDim.x >> 5) + (threadIdx.x >> 5);
  if (w >= ND) return;
  const float* p = drug + (size_t)w * 128;
  float s = 0.0f;
  for (int i = lane; i < 128; i += 32) s += p[i] * qk[i];
#pragma unroll
  for (int d = 16; d; d >>= 1) s += __shfl_xor(s, d, 32);
  if (lane == 0) dscore[w] = (s + ckp[0]) * 0.08838834764831845f; // 1/sqrt(128)
}

__global__ void zero_i_k(int* p, int n) {
  int i = blockIdx.x * blockDim.x + threadIdx.x;
  if (i < n) p[i] = 0;
}

__global__ void invperm_k(const int* __restrict__ nb, int* __restrict__ inv, int N) {
  int i = blockIdx.x * blockDim.x + threadIdx.x;
  if (i < N) inv[nb[i]] = i;
}

__global__ void npos_count_k(const int* __restrict__ src, const int* __restrict__ inv,
                             int* __restrict__ npos, int* __restrict__ cnt, int E) {
  int e = blockIdx.x * blockDim.x + threadIdx.x;
  if (e < E) {
    int n = inv[src[e]];
    npos[e] = n;
    atomicAdd(&cnt[n], 1);
  }
}

// Single-block exclusive scan of cnt[N] -> off[N] (and cursor copy)
__global__ void scan_k(const int* __restrict__ cnt, int* __restrict__ off,
                       int* __restrict__ cursor, int N)
{
  __shared__ int sums[1024];
  const int t = threadIdx.x;
  const int chunk = (N + 1023) / 1024;
  const int beg = t * chunk;
  const int end = min(beg + chunk, N);
  int s = 0;
  for (int i = beg; i < end; ++i) s += cnt[i];
  sums[t] = s;
  __syncthreads();
  for (int d = 1; d < 1024; d <<= 1) {
    int v = (t >= d) ? sums[t - d] : 0;
    __syncthreads();
    sums[t] += v;
    __syncthreads();
  }
  int run = (t > 0) ? sums[t - 1] : 0;
  for (int i = beg; i < end; ++i) {
    off[i] = run;
    cursor[i] = run;
    run += cnt[i];
  }
}

__global__ void fill_k(const int* __restrict__ npos, int* __restrict__ cursor,
                       int* __restrict__ elist, int E) {
  int e = blockIdx.x * blockDim.x + threadIdx.x;
  if (e < E) {
    int p = atomicAdd(&cursor[npos[e]], 1);
    elist[p] = e;
  }
}

// One wave per node: segment softmax over its edges, then accumulate
//   G[n] = sum_e attn_e*coeff_e*drug[drug_idx[e]]  (written bf16 -> msgs GEMM A)
//   svec[n] = sum_e attn_e*coeff_e
__global__ void node_k(const int* __restrict__ off, const int* __restrict__ cnt,
                       const int* __restrict__ elist, const int* __restrict__ didx,
                       const float* __restrict__ yv, const float* __restrict__ wv,
                       const float* __restrict__ dscore, const float* __restrict__ drug,
                       __bf16* __restrict__ Gb, float* __restrict__ svec, int N)
{
  const int lane = threadIdx.x & 31;
  const int node = blockIdx.x * (blockDim.x >> 5) + (threadIdx.x >> 5);
  if (node >= N) return;
  const int beg = off[node];
  const int c   = cnt[node];
  float4 acc = make_float4(0.0f, 0.0f, 0.0f, 0.0f);
  float ssum = 0.0f;
  if (c > 0) {
    float mx = -1e30f;
    for (int i = lane; i < c; i += 32)
      mx = fmaxf(mx, dscore[didx[elist[beg + i]]]);
#pragma unroll
    for (int d = 16; d; d >>= 1) mx = fmaxf(mx, __shfl_xor(mx, d, 32));
    float se = 0.0f;
    for (int i = lane; i < c; i += 32)
      se += __expf(dscore[didx[elist[beg + i]]] - mx);
#pragma unroll
    for (int d = 16; d; d >>= 1) se += __shfl_xor(se, d, 32);
    const float inv = 1.0f / (se + 1e-12f);
    for (int i = 0; i < c; ++i) {
      const int e  = elist[beg + i];
      const int dj = didx[e];
      const float we = __expf(dscore[dj] - mx) * inv * ((yv[e] - 6.0f) * wv[e]);
      const float4 dv = reinterpret_cast<const float4*>(drug + (size_t)dj * 128)[lane];
      acc.x += we * dv.x; acc.y += we * dv.y;
      acc.z += we * dv.z; acc.w += we * dv.w;
      ssum += we;
    }
  }
  union { __bf16 h[4]; uint2 u; } pk;
  pk.h[0] = (bf16_t)acc.x; pk.h[1] = (bf16_t)acc.y;
  pk.h[2] = (bf16_t)acc.z; pk.h[3] = (bf16_t)acc.w;
  *reinterpret_cast<uint2*>(Gb + (size_t)node * 128 + 4 * lane) = pk.u;
  if (lane == 0) svec[node] = ssum;
}

// attn[n] = A2[n,0:64] . W3 + b3   (A2 stored bf16)
__global__ void attn3_k(const __bf16* __restrict__ A2, const float* __restrict__ W3,
                        const float* __restrict__ b3p, float* __restrict__ attn, int N) {
  int n = blockIdx.x * blockDim.x + threadIdx.x;
  if (n < N) {
    float s = b3p[0];
    const __bf16* p = A2 + (size_t)n * 64;
    for (int i = 0; i < 64; ++i) s += (float)p[i] * W3[i];
    attn[n] = s;
  }
}

__global__ void amax_k(const float* __restrict__ af, const float* __restrict__ ar,
                       int N, float* __restrict__ outmax) {
  const float* x = blockIdx.x ? ar : af;
  __shared__ float sm[1024];
  float m = -1e30f;
  for (int i = threadIdx.x; i < N; i += 1024) m = fmaxf(m, x[i]);
  sm[threadIdx.x] = m;
  __syncthreads();
  for (int s = 512; s; s >>= 1) {
    if (threadIdx.x < s) sm[threadIdx.x] = fmaxf(sm[threadIdx.x], sm[threadIdx.x + s]);
    __syncthreads();
  }
  if (threadIdx.x == 0) outmax[blockIdx.x] = sm[0];
}

__global__ void esum_k(const float* __restrict__ af, const float* __restrict__ ar,
                       const float* __restrict__ mx, float* __restrict__ es, int N) {
  const float* x = blockIdx.x ? ar : af;
  const float m = mx[blockIdx.x];
  __shared__ float sm[1024];
  float s = 0.0f;
  for (int i = threadIdx.x; i < N; i += 1024) s += __expf(x[i] - m);
  sm[threadIdx.x] = s;
  __syncthreads();
  for (int k = 512; k; k >>= 1) {
    if (threadIdx.x < k) sm[threadIdx.x] += sm[threadIdx.x + k];
    __syncthreads();
  }
  if (threadIdx.x == 0) es[blockIdx.x] = sm[0];
}

__global__ void softmax_out_k(const float* __restrict__ af, const float* __restrict__ ar,
                              const float* __restrict__ mx, const float* __restrict__ es,
                              float* __restrict__ osf, float* __restrict__ osr, int N) {
  int i = blockIdx.x * blockDim.x + threadIdx.x;
  if (i < N)            osf[i]     = __expf(af[i] - mx[0]) / es[0];
  else if (i < 2 * N)   osr[i - N] = __expf(ar[i - N] - mx[1]) / es[1];
}

// v_prior[c] = sum_n softmax_joint(a)[n] * msgs[n][c]  (single block: deterministic)
__global__ void vprior_k(const float* __restrict__ af, const float* __restrict__ ar,
                         const float* __restrict__ mx, const float* __restrict__ es,
                         const float* __restrict__ msgsF, const float* __restrict__ msgsR,
                         float* __restrict__ vpa, int N) {
  const int c = threadIdx.x;                 // 128 threads
  const float mj  = fmaxf(mx[0], mx[1]);
  const float inv = 1.0f / (es[0] * __expf(mx[0] - mj) + es[1] * __expf(mx[1] - mj));
  float acc = 0.0f;
  for (int n = 0; n < N; ++n) acc += __expf(af[n] - mj) * msgsF[(size_t)n * 128 + c];
  for (int n = 0; n < N; ++n) acc += __expf(ar[n] - mj) * msgsR[(size_t)n * 128 + c];
  vpa[c] = acc * inv;
}

// Integrator MLP + residual + LayerNorm -> out[0:256]. Single block of 256.
__global__ void final_k(const float* __restrict__ vpa,
                        const float* iW1, const float* ib1, const float* ia1,
                        const float* iW2, const float* ib2,
                        const float* lng, const float* lnb,
                        const float* __restrict__ zt, float* __restrict__ out) {
  __shared__ float vp[128], h[256], red[256];
  const int t = threadIdx.x;
  if (t < 128) vp[t] = vpa[t];
  __syncthreads();
  float s = ib1[t];
  for (int i = 0; i < 128; ++i) s += iW1[t * 128 + i] * vp[i];
  const float a = ia1[0];
  h[t] = (s >= 0.0f) ? s : a * s;
  __syncthreads();
  s = ib2[t];
  for (int i = 0; i < 256; ++i) s += iW2[t * 256 + i] * h[i];
  const float x = zt[t] + s;
  red[t] = x;
  __syncthreads();
  for (int k = 128; k; k >>= 1) { if (t < k) red[t] += red[t + k]; __syncthreads(); }
  const float mean = red[0] / 256.0f;
  __syncthreads();
  const float d = x - mean;
  red[t] = d * d;
  __syncthreads();
  for (int k = 128; k; k >>= 1) { if (t < k) red[t] += red[t + k]; __syncthreads(); }
  const float var = red[0] / 256.0f;
  out[t] = d * rsqrtf(var + 1e-5f) * lng[t] + lnb[t];
}

// ============================================================================
static inline int gemm_blocks(int M, int N) {
  const int waves = ((M + 15) / 16) * (N / 64);
  return (waves + 7) / 8;
}

extern "C" void kernel_launch(void* const* d_in, const int* in_sizes, int n_in,
                              void* d_out, int out_size, void* d_ws, size_t ws_size,
                              hipStream_t stream)
{
  (void)in_sizes; (void)n_in; (void)out_size; (void)ws_size;
  constexpr int NF = 20000, NR = 20000, E = 500000, ND = 50000;

  // ---- inputs (setup_inputs dict order, params flattened in key order) ----
  const float* target   = (const float*)d_in[0];
  const float* formX    = (const float*)d_in[1];
  const float* roleX    = (const float*)d_in[2];
  const int*   formNb   = (const int*)d_in[3];
  const int*   formSrc  = (const int*)d_in[4];
  const int*   formDrug = (const int*)d_in[5];
  const float* formY    = (const float*)d_in[6];
  const float* formWt   = (const float*)d_in[7];
  const int*   roleNb   = (const int*)d_in[8];
  const int*   roleSrc  = (const int*)d_in[9];
  const int*   roleDrug = (const int*)d_in[10];
  const float* roleY    = (const float*)d_in[11];
  const float* roleWt   = (const float*)d_in[12];
  const float* drugF    = (const float*)d_in[13];
  const float* fW1 = (const float*)d_in[14];
  const float* fb1 = (const float*)d_in[15];
  const float* fa1 = (const float*)d_in[16];
  const float* fW2 = (const float*)d_in[17];
  const float* fb2 = (const float*)d_in[18];
  const float* rW1 = (const float*)d_in[19];
  const float* rb1 = (const float*)d_in[20];
  const float* ra1 = (const float*)d_in[21];
  const float* rW2 = (const float*)d_in[22];
  const float* rb2 = (const float*)d_in[23];
  const float* qW  = (const float*)d_in[24];
  const float* qb  = (const float*)d_in[25];
  const float* kW  = (const float*)d_in[26];
  const float* kb  = (const float*)d_in[27];
  const float* vW  = (const float*)d_in[28];
  const float* vb  = (const float*)d_in[29];
  const float* lemb = (const float*)d_in[30];
  const float* aW1  = (const float*)d_in[31];
  const float* ab1  = (const float*)d_in[32];
  const float* aW2  = (const float*)d_in[33];
  const float* ab2  = (const float*)d_in[34];
  const float* aW3  = (const float*)d_in[35];
  const float* ab3  = (const float*)d_in[36];
  const float* iW1  = (const float*)d_in[37];
  const float* ib1  = (const float*)d_in[38];
  const float* ia1  = (const float*)d_in[39];
  const float* iW2  = (const float*)d_in[40];
  const float* ib2  = (const float*)d_in[41];
  const float* lng  = (const float*)d_in[42];
  const float* lnb  = (const float*)d_in[43];

  // ---- outputs: z_refined[256] | form_msgs | role_msgs | sm(form) | sm(role)
  float* out   = (float*)d_out;
  float* outz  = out;
  float* msgsF = out + 256;
  float* msgsR = out + 256 + (size_t)NF * 128;
  float* osf   = out + 256 + (size_t)(NF + NR) * 128;
  float* osr   = osf + NF;

  // ---- workspace carve-up ----
  char* base = (char*)d_ws;
  size_t off = 0;
  auto alloc = [&](size_t bytes) -> void* {
    off = (off + 511) & ~(size_t)511;
    void* p = base + off;
    off += bytes;
    return p;
  };
  float*  ztg    = (float*)alloc(256 * sizeof(float));
  float*  qk     = (float*)alloc(128 * sizeof(float));
  float*  ck     = (float*)alloc(8   * sizeof(float));
  float*  c0f    = (float*)alloc(128 * sizeof(float));
  float*  c0r    = (float*)alloc(128 * sizeof(float));
  float*  dscore = (float*)alloc((size_t)ND * sizeof(float));
  float*  mx2    = (float*)alloc(8 * sizeof(float));
  float*  es2    = (float*)alloc(8 * sizeof(float));
  float*  vpa    = (float*)alloc(128 * sizeof(float));
  float*  svec   = (float*)alloc((size_t)NF * sizeof(float));
  float*  af     = (float*)alloc((size_t)NF * sizeof(float));
  float*  ar     = (float*)alloc((size_t)NR * sizeof(float));
  int*    inv    = (int*)alloc((size_t)NF * sizeof(int));
  int*    cnt    = (int*)alloc((size_t)NF * sizeof(int));
  int*    offn   = (int*)alloc((size_t)NF * sizeof(int));
  int*    cursor = (int*)alloc((size_t)NF * sizeof(int));
  int*    npos   = (int*)alloc((size_t)E * sizeof(int));
  int*    elist  = (int*)alloc((size_t)E * sizeof(int));
  // bf16 operand pools
  __bf16* Xfb    = (__bf16*)alloc((size_t)NF * 256 * sizeof(__bf16));
  __bf16* Xrb    = (__bf16*)alloc((size_t)NR * 256 * sizeof(__bf16));
  __bf16* BIGa   = (__bf16*)alloc((size_t)NF * 256 * sizeof(__bf16)); // H / A1 / G
  __bf16* BIGb   = (__bf16*)alloc((size_t)NF * 256 * sizeof(__bf16)); // R / A2
  __bf16* fW1b   = (__bf16*)alloc(256 * 256 * sizeof(__bf16));
  __bf16* fW2b   = (__bf16*)alloc(256 * 256 * sizeof(__bf16));
  __bf16* rW1b   = (__bf16*)alloc(256 * 256 * sizeof(__bf16));
  __bf16* rW2b   = (__bf16*)alloc(256 * 256 * sizeof(__bf16));
  __bf16* vWb    = (__bf16*)alloc(128 * 128 * sizeof(__bf16));
  __bf16* aW2b   = (__bf16*)alloc(64 * 128 * sizeof(__bf16));
  __bf16* aW1zb  = (__bf16*)alloc(128 * 256 * sizeof(__bf16)); // attn_W1[:,256:512]
  __bf16* Gb     = BIGa;                                        // [NF,128] (pre-GEMM1)

  // ---- one-time bf16 conversions (weights + node features) ----
  auto cvt = [&](const float* src, __bf16* dst, int n) {
    cvt_k<<<(n + 255) / 256, 256, 0, stream>>>(src, dst, n);
  };
  cvt(formX, Xfb, NF * 256);
  cvt(roleX, Xrb, NR * 256);
  cvt(fW1, fW1b, 256 * 256);
  cvt(fW2, fW2b, 256 * 256);
  cvt(rW1, rW1b, 256 * 256);
  cvt(rW2, rW2b, 256 * 256);
  cvt(vW,  vWb,  128 * 128);
  cvt(aW2, aW2b, 64 * 128);
  cvt_stride_k<<<(128 * 256 + 255) / 256, 256, 0, stream>>>(aW1, aW1zb, 128, 528, 256, 256);

  // ---- setup: z_t, folded score vector qk/ck, attn layer-1 const biases ----
  setup_k<<<1, 256, 0, stream>>>(target, fW1, fb1, fa1, fW2, fb2,
                                 rW1, rb1, ra1, rW2, rb2,
                                 qW, qb, kW, kb, lemb, aW1, ab1,
                                 ztg, qk, ck, c0f, c0r);
  dscore_k<<<(ND + 7) / 8, 256, 0, stream>>>(drugF, qk, ck, dscore, ND);

  // ---- per-graph pipeline (form then role; buffers reused) ----
  auto run_graph = [&](const int* nb, const int* src, const int* dg,
                       const float* yv, const float* wv, const __bf16* Xb,
                       const __bf16* W1b, const float* b1, const float* a1,
                       const __bf16* W2b, const float* b2, const float* c0,
                       float* msgs_out, float* attn_out) {
    const int eb = (E + 255) / 256;
    zero_i_k    <<<(NF + 255) / 256, 256, 0, stream>>>(cnt, NF);
    invperm_k   <<<(NF + 255) / 256, 256, 0, stream>>>(nb, inv, NF);
    npos_count_k<<<eb, 256, 0, stream>>>(src, inv, npos, cnt, E);
    scan_k      <<<1, 1024, 0, stream>>>(cnt, offn, cursor, NF);
    fill_k      <<<eb, 256, 0, stream>>>(npos, cursor, elist, E);
    node_k      <<<(NF + 7) / 8, 256, 0, stream>>>(offn, cnt, elist, dg, yv, wv,
                                                   dscore, drugF, Gb, svec, NF);
    // msgs = G @ v_W^T + svec[m]*v_b[n]   (WMMA, f32 out -> d_out)
    gemm_k<128, 0, false><<<gemm_blocks(NF, 128), 256, 0, stream>>>(
        Gb, 128, vWb, 128, nullptr, svec, vb, msgs_out, nullptr, 128, NF, 128, nullptr);
    // refiner: H = prelu(X@W1^T+b1, a1); R = H@W2^T+b2   (WMMA, bf16 chain)
    gemm_k<256, 1, true><<<gemm_blocks(NF, 256), 256, 0, stream>>>(
        Xb, 256, W1b, 256, b1, nullptr, nullptr, nullptr, BIGa, 256, NF, 256, a1);
    gemm_k<256, 0, true><<<gemm_blocks(NF, 256), 256, 0, stream>>>(
        BIGa, 256, W2b, 256, b2, nullptr, nullptr, nullptr, BIGb, 256, NF, 256, nullptr);
    // attn layer1 folded: A1 = leaky(R @ W1z^T + c0)
    gemm_k<256, 2, true><<<gemm_blocks(NF, 128), 256, 0, stream>>>(
        BIGb, 256, aW1zb, 256, c0, nullptr, nullptr, nullptr, BIGa, 128, NF, 128, nullptr);
    // attn layer2: A2 = leaky(A1 @ attn_W2^T + attn_b2)
    gemm_k<128, 2, true><<<gemm_blocks(NF, 64), 256, 0, stream>>>(
        BIGa, 128, aW2b, 128, ab2, nullptr, nullptr, nullptr, BIGb, 64, NF, 64, nullptr);
    attn3_k<<<(NF + 255) / 256, 256, 0, stream>>>(BIGb, aW3, ab3, attn_out, NF);
  };

  run_graph(formNb, formSrc, formDrug, formY, formWt, Xfb,
            fW1b, fb1, fa1, fW2b, fb2, c0f, msgsF, af);
  run_graph(roleNb, roleSrc, roleDrug, roleY, roleWt, Xrb,
            rW1b, rb1, ra1, rW2b, rb2, c0r, msgsR, ar);

  // ---- softmaxes, v_prior, integrator + layernorm ----
  amax_k<<<2, 1024, 0, stream>>>(af, ar, NF, mx2);
  esum_k<<<2, 1024, 0, stream>>>(af, ar, mx2, es2, NF);
  softmax_out_k<<<(2 * NF + 255) / 256, 256, 0, stream>>>(af, ar, mx2, es2, osf, osr, NF);
  vprior_k<<<1, 128, 0, stream>>>(af, ar, mx2, es2, msgsF, msgsR, vpa, NF);
  final_k<<<1, 256, 0, stream>>>(vpa, iW1, ib1, ia1, iW2, ib2, lng, lnb, ztg, outz);
}
